// GravityAttention_12446815224462
// MI455X (gfx1250) — matrix operand order repaired and checked by reading, not compiled, not run on hardware
//
#include <hip/hip_runtime.h>

// ---------------------------------------------------------------------------
// GravityAttention for MI455X (gfx1250, wave32, WMMA + TDM).
// B=2, S=2048, D=1024, H=16, DH=64, CD=DC=16.
// Mixed precision: bf16 WMMA operands / fp32 accumulate; conversions hoisted
// out of all hot loops; V tiles staged to LDS via the Tensor Data Mover.
// ---------------------------------------------------------------------------

#define B_DIM 2
#define S_LEN 2048
#define D_DIM 1024
#define H_NUM 16
#define DH_DIM 64
#define DC_DIM 16

typedef __attribute__((ext_vector_type(16))) __bf16 v16bf;
typedef __attribute__((ext_vector_type(8)))  float  v8f;
typedef __attribute__((ext_vector_type(2)))  float  v2f;
typedef __attribute__((ext_vector_type(8)))  unsigned short us8;
typedef __attribute__((ext_vector_type(4)))  unsigned int   ui4;
typedef __attribute__((ext_vector_type(8)))  int            i8v;
typedef __attribute__((ext_vector_type(4)))  int            i4v;

union BF16x16 { v16bf v; us8 h[2]; unsigned short u[16]; };

__device__ __forceinline__ unsigned short f2bf(float f) {
    union { __bf16 b; unsigned short u; } c;
    c.b = (__bf16)f;                          // native v_cvt on gfx1250
    return c.u;
}

// ---------------------------------------------------------------------------
// Bulk f32 -> bf16 conversion (one pass, outside all hot loops).
// ---------------------------------------------------------------------------
__global__ __launch_bounds__(256) void cvt_f32_bf16(
    const float* __restrict__ src, unsigned short* __restrict__ dst, int n)
{
    const int i = (blockIdx.x * 256 + threadIdx.x) * 4;
    if (i + 3 < n) {
#pragma unroll
        for (int j = 0; j < 4; j++) dst[i + j] = f2bf(src[i + j]);
    }
}

// ---------------------------------------------------------------------------
// C = Xb * Wb^T + bias   (Xb: [M,K] bf16, Wb: [N,K] bf16)
// Inner loop: pure b128 loads + v_wmma_f32_16x16x32_bf16, fp32 accumulate.
// Block = 256 threads = 8 waves; block tile 128(M) x 64(N); wave: 16 x 64.
// ---------------------------------------------------------------------------
template <bool OUT_BF16>
__global__ __launch_bounds__(256) void gemm_bf16(
    const unsigned short* __restrict__ Xb, const unsigned short* __restrict__ Wb,
    const float* __restrict__ bias, void* __restrict__ Cout,
    int M, int N, int K)
{
    const int wave = threadIdx.x >> 5;
    const int lane = threadIdx.x & 31;
    const int half = lane >> 4;
    const int l16  = lane & 15;

    const int m0 = blockIdx.x * 128 + wave * 16;
    const int n0 = blockIdx.y * 64;

    v8f acc[4] = {};

    for (int k0 = 0; k0 < K; k0 += 32) {
        // A fragment: row = m0+l16; K elems {kb..kb+15}, kb=16*half
        const unsigned short* arow = Xb + (size_t)(m0 + l16) * K + k0 + half * 16;
        BF16x16 a;
        a.h[0] = *(const us8*)(arow);
        a.h[1] = *(const us8*)(arow + 8);
#pragma unroll
        for (int nt = 0; nt < 4; nt++) {
            // B fragment: B[k][n] = W[n][k]; n = n0+nt*16+l16, k = k0+16*half+j
            const unsigned short* wrow =
                Wb + (size_t)(n0 + nt * 16 + l16) * K + k0 + half * 16;
            BF16x16 b;
            b.h[0] = *(const us8*)(wrow);
            b.h[1] = *(const us8*)(wrow + 8);
            acc[nt] = __builtin_amdgcn_wmma_f32_16x16x32_bf16(
                false, a.v, false, b.v, (short)0, acc[nt], false, false);
        }
    }

    // D layout: row m = m0 + half*8 + r, col n = n0 + nt*16 + l16
#pragma unroll
    for (int nt = 0; nt < 4; nt++) {
        const int n = n0 + nt * 16 + l16;
        const float bia = bias[n];
#pragma unroll
        for (int r = 0; r < 8; r++) {
            const size_t idx = (size_t)(m0 + half * 8 + r) * N + n;
            if constexpr (OUT_BF16)
                ((unsigned short*)Cout)[idx] = f2bf(acc[nt][r] + bia);
            else
                ((float*)Cout)[idx] = acc[nt][r] + bia;
        }
    }
}

// ---------------------------------------------------------------------------
// z[b,h,s,c] = coords[b,s,:] . Wc[h*16+c,:] + bc   (tiny GEMM, VALU)
// ---------------------------------------------------------------------------
__global__ void coord_proj(const float* __restrict__ coords,
                           const float* __restrict__ Wc,
                           const float* __restrict__ bc,
                           float* __restrict__ z)
{
    const int tid = blockIdx.x * 256 + threadIdx.x;      // B*S*H*DC = 1048576
    const int bs  = tid >> 8;                            // b*S + s
    const int hc  = tid & 255;                           // h*16 + c
    float acc = bc[hc];
#pragma unroll
    for (int cc = 0; cc < DC_DIM; cc++)
        acc += coords[(size_t)bs * DC_DIM + cc] * Wc[(size_t)hc * DC_DIM + cc];
    const int b = bs / S_LEN, s = bs % S_LEN;
    const int h = hc >> 4,   c = hc & 15;
    z[(((size_t)(b * H_NUM + h) * S_LEN) + s) * DC_DIM + c] = acc;
}

// zsq[b,h,s] = sum_c z^2
__global__ void zsq_kernel(const float* __restrict__ z, float* __restrict__ zsq)
{
    const int tid = blockIdx.x * 256 + threadIdx.x;      // 65536
    float acc = 0.f;
#pragma unroll
    for (int c = 0; c < DC_DIM; c++) {
        float v = z[(size_t)tid * DC_DIM + c];
        acc += v * v;
    }
    zsq[tid] = acc;
}

// updated_coords[b,s,c] = hidden[b,s,:] . Wn[c,:] + bn[c]
__global__ void newcoords_kernel(const float* __restrict__ hidden,
                                 const float* __restrict__ Wn,
                                 const float* __restrict__ bn,
                                 float* __restrict__ outc)
{
    const int tid = blockIdx.x * 256 + threadIdx.x;      // 4096*16
    const int bs = tid >> 4;
    const int c  = tid & 15;
    float acc = bn[c];
    const float* hrow = hidden + (size_t)bs * D_DIM;
    const float* wrow = Wn + (size_t)c * D_DIM;
    for (int k = 0; k < D_DIM; k += 4) {
        acc += hrow[k]     * wrow[k];
        acc += hrow[k + 1] * wrow[k + 1];
        acc += hrow[k + 2] * wrow[k + 2];
        acc += hrow[k + 3] * wrow[k + 3];
    }
    outc[tid] = acc;
}

// ---------------------------------------------------------------------------
// Flash-style gravity attention.
// grid = (S/128, H, B); 256 threads = 8 waves; each wave owns 16 query rows.
// Per 32-key chunk:
//   V tile (32x64 bf16, global row pitch 1024) staged LDS-ward by the TDM
//   Sᵀ tiles (key=M, query=N) via 2 x (4 chained v_wmma_f32_16x16x4_f32)
//   online softmax (stats per query live in lane pairs -> shfl_xor 16)
//   P bf16 A-frag built directly from the two Sᵀ D-tiles (no transpose)
//   out += P*V via 4 x v_wmma_f32_16x16x32_bf16
// Output written directly as bf16 (consumed only as GEMM2 A operand).
// ---------------------------------------------------------------------------
__global__ __launch_bounds__(256) void gravity_attn(
    const float* __restrict__ z, const float* __restrict__ zsq,
    const unsigned short* __restrict__ Vbf, const float* __restrict__ gamma,
    unsigned short* __restrict__ Obf)
{
    __shared__ float          lds_zq[128 * DC_DIM];
    __shared__ float          lds_zqsq[128];
    __shared__ float          lds_zk[32 * DC_DIM];
    __shared__ float          lds_zksq[32];
    __shared__ unsigned short lds_vb[32 * DH_DIM];   // bf16 V chunk (TDM dest)

    const int bb = blockIdx.z;
    const int h  = blockIdx.y;
    const int q0 = blockIdx.x * 128;

    const float* zh   = z   + (size_t)(bb * H_NUM + h) * S_LEN * DC_DIM;
    const float* zsqh = zsq + (size_t)(bb * H_NUM + h) * S_LEN;

    const float g = log1pf(__expf(gamma[h]));        // softplus

    const int wave = threadIdx.x >> 5;
    const int lane = threadIdx.x & 31;
    const int half = lane >> 4;
    const int l16  = lane & 15;
    const int qr0  = wave * 16;                      // wave's query rows in block

    // stage query tile
    for (int i = threadIdx.x; i < 128 * DC_DIM; i += 256)
        lds_zq[i] = zh[(size_t)q0 * DC_DIM + i];
    for (int i = threadIdx.x; i < 128; i += 256)
        lds_zqsq[i] = zsqh[q0 + i];
    __syncthreads();

    // B fragments of z_qᵀ (loop invariant): B[c][s]=zq[s][c], c = kk*4+2*half+e
    v2f bq[4];
#pragma unroll
    for (int kk = 0; kk < 4; kk++) {
        bq[kk].x = lds_zq[(qr0 + l16) * DC_DIM + kk * 4 + half * 2 + 0];
        bq[kk].y = lds_zq[(qr0 + l16) * DC_DIM + kk * 4 + half * 2 + 1];
    }
    const float qs = lds_zqsq[qr0 + l16];            // |z_q|^2 for query l16

    float m_i = -3.0e38f, l_i = 0.0f;
    v8f oacc[4] = {};

    for (int t0 = 0; t0 < S_LEN; t0 += 32) {
        __syncthreads();
        for (int i = threadIdx.x; i < 32 * DC_DIM; i += 256)
            lds_zk[i] = zh[(size_t)t0 * DC_DIM + i];
        for (int i = threadIdx.x; i < 32; i += 256)
            lds_zksq[i] = zsqh[t0 + i];

        // ---- V chunk via Tensor Data Mover: 32(rows) x 64(bf16) tile,
        //      global row stride 1024 elems -> contiguous LDS tile ----------
        if (wave == 0) {                             // one TDM issue per block
            const unsigned lds_addr = (unsigned)(uintptr_t)(void*)lds_vb;
            const unsigned long long ga = (unsigned long long)(uintptr_t)
                (const void*)&Vbf[((size_t)bb * S_LEN + t0) * D_DIM + h * DH_DIM];
            ui4 g0;
            g0.x = 1u;                                         // count=1 (valid)
            g0.y = lds_addr;                                   // lds_addr[31:0]
            g0.z = (unsigned)(ga & 0xffffffffull);             // global_addr lo
            g0.w = (unsigned)((ga >> 32) & 0x01ffffffull)      // global_addr[56:32]
                 | (2u << 30);                                 // type = 2 (image)
            i8v g1;
            g1[0] = (1 << 16);        // workgroup_mask=0, data_size=1 (2 bytes)
            g1[1] = (64 << 16);       // tensor_dim0 = 64      (bits 63:48)
            g1[2] = (32 << 16);       // tensor_dim1 = 32      (bits 95:80)
            g1[3] = (64 << 16);       // tile_dim0   = 64      (bits 127:112)
            g1[4] = 32;               // tile_dim1 = 32, tile_dim2 = 0
            g1[5] = D_DIM;            // tensor_dim0_stride = 1024 (bits 191:160)
            g1[6] = 0;
            g1[7] = 0;
            i4v gz = {0, 0, 0, 0};
#if __clang_major__ >= 23
            i8v gz8 = {0, 0, 0, 0, 0, 0, 0, 0};
            __builtin_amdgcn_tensor_load_to_lds(g0, g1, gz, gz, gz8, 0);
#else
            __builtin_amdgcn_tensor_load_to_lds(g0, g1, gz, gz, 0);
#endif
            __builtin_amdgcn_s_wait_tensorcnt(0);
        }
        if (t0 + 32 < S_LEN)                         // prefetch next z chunk
            __builtin_prefetch(&zh[(size_t)(t0 + 32) * DC_DIM + threadIdx.x], 0, 3);
        __syncthreads();

        // ---- Sᵀ = z_k . z_qᵀ, two 16-key subtiles, exact f32 WMMA ----
        float sv0[8], sv1[8];
#pragma unroll
        for (int th = 0; th < 2; th++) {
            v8f sacc = {};
#pragma unroll
            for (int kk = 0; kk < 4; kk++) {
                v2f ak;                              // A row = key l16 of subtile
                ak.x = lds_zk[(th * 16 + l16) * DC_DIM + kk * 4 + half * 2 + 0];
                ak.y = lds_zk[(th * 16 + l16) * DC_DIM + kk * 4 + half * 2 + 1];
                sacc = __builtin_amdgcn_wmma_f32_16x16x4_f32(
                    false, ak, false, bq[kk], (short)0, sacc, false, false);
            }
            float* sv = th ? sv1 : sv0;
#pragma unroll
            for (int r = 0; r < 8; r++) {            // key = t0+th*16+half*8+r
                float dist = qs + lds_zksq[th * 16 + half * 8 + r] - 2.0f * sacc[r];
                sv[r] = -g * fmaxf(dist, 0.0f);
            }
        }

        // ---- online softmax (per-query stats in lane pair l16 / l16+16) ----
        float cm = -3.0e38f;
#pragma unroll
        for (int r = 0; r < 8; r++) cm = fmaxf(cm, fmaxf(sv0[r], sv1[r]));
        cm = fmaxf(cm, __shfl_xor(cm, 16));
        const float mnew = fmaxf(m_i, cm);
        float cs = 0.0f;
#pragma unroll
        for (int r = 0; r < 8; r++) {
            sv0[r] = __expf(sv0[r] - mnew);
            sv1[r] = __expf(sv1[r] - mnew);
            cs += sv0[r] + sv1[r];
        }
        cs += __shfl_xor(cs, 16);
        const float scale = __expf(m_i - mnew);
        l_i = l_i * scale + cs;
        m_i = mnew;
#pragma unroll
        for (int r = 0; r < 8; r++) {                // rescale acc rows half*8+r
            const float fr = __shfl(scale, half * 8 + r);
            oacc[0][r] *= fr; oacc[1][r] *= fr; oacc[2][r] *= fr; oacc[3][r] *= fr;
        }

        // ---- P A-fragment: direct from Sᵀ tiles (k = half*8 + j / +16) ----
        BF16x16 pa;
#pragma unroll
        for (int r = 0; r < 8; r++) { pa.u[r] = f2bf(sv0[r]); pa.u[r + 8] = f2bf(sv1[r]); }

#pragma unroll
        for (int nt = 0; nt < 4; nt++) {             // V B-frag: k=8*half+(j&7)+16*(j>>3), d=nt*16+l16
            BF16x16 vb;
#pragma unroll
            for (int j = 0; j < 16; j++)
                vb.u[j] = lds_vb[(half * 8 + (j & 7) + (j >> 3) * 16) * DH_DIM + nt * 16 + l16];
            oacc[nt] = __builtin_amdgcn_wmma_f32_16x16x32_bf16(
                false, pa.v, false, vb.v, (short)0, oacc[nt], false, false);
        }
    }

    // finalize: divide row q by l[q], write bf16 [B,S,D] (GEMM2 A operand)
#pragma unroll
    for (int r = 0; r < 8; r++) {
        const float lr  = __shfl(l_i, half * 8 + r);
        const float inv = 1.0f / lr;
        const int qg = q0 + qr0 + half * 8 + r;
#pragma unroll
        for (int nt = 0; nt < 4; nt++)
            Obf[((size_t)bb * S_LEN + qg) * D_DIM + h * DH_DIM + nt * 16 + l16] =
                f2bf(oacc[nt][r] * inv);
    }
}

// ---------------------------------------------------------------------------
extern "C" void kernel_launch(void* const* d_in, const int* in_sizes, int n_in,
                              void* d_out, int out_size, void* d_ws, size_t ws_size,
                              hipStream_t stream)
{
    const float* hidden = (const float*)d_in[0];
    const float* coords = (const float*)d_in[1];
    const float* Wv     = (const float*)d_in[2];
    const float* bv     = (const float*)d_in[3];
    const float* Wc     = (const float*)d_in[4];
    const float* bc     = (const float*)d_in[5];
    const float* Wn     = (const float*)d_in[6];
    const float* bn     = (const float*)d_in[7];
    const float* Wo     = (const float*)d_in[8];
    const float* bo     = (const float*)d_in[9];
    const float* gamma  = (const float*)d_in[10];

    float* out_hidden = (float*)d_out;                                  // [B,S,D]
    float* out_coords = out_hidden + (size_t)B_DIM * S_LEN * D_DIM;     // [B,S,CD]

    const size_t nBSD = (size_t)B_DIM * S_LEN * D_DIM;                  // 4194304
    const size_t nDD  = (size_t)D_DIM * D_DIM;                          // 1048576

    // f32 scratch first (alignment), then bf16 scratch
    float* zbuf   = (float*)d_ws;                                       // 1M f32
    float* zsqbuf = zbuf + (size_t)B_DIM * H_NUM * S_LEN * DC_DIM;      // 64K f32
    unsigned short* Xbf  = (unsigned short*)(zsqbuf + (size_t)B_DIM * H_NUM * S_LEN);
    unsigned short* Wvbf = Xbf  + nBSD;
    unsigned short* Wobf = Wvbf + nDD;
    unsigned short* Vbf  = Wobf + nDD;
    unsigned short* Obf  = Vbf  + nBSD;

    // one-time bf16 conversions (hoisted out of all GEMM loops)
    cvt_f32_bf16<<<(int)(nBSD / 1024), 256, 0, stream>>>(hidden, Xbf, (int)nBSD);
    cvt_f32_bf16<<<(int)(nDD  / 1024), 256, 0, stream>>>(Wv, Wvbf, (int)nDD);
    cvt_f32_bf16<<<(int)(nDD  / 1024), 256, 0, stream>>>(Wo, Wobf, (int)nDD);

    // value projection: V = hidden @ Wv^T + bv  (bf16 out: WMMA-B operand only)
    gemm_bf16<true><<<dim3(32, 16), 256, 0, stream>>>(
        Xbf, Wvbf, bv, (void*)Vbf, B_DIM * S_LEN, D_DIM, D_DIM);

    // coordinate heads + |z|^2
    coord_proj<<<4096, 256, 0, stream>>>(coords, Wc, bc, zbuf);
    zsq_kernel<<<256, 256, 0, stream>>>(zbuf, zsqbuf);

    // updated coordinates (independent path)
    newcoords_kernel<<<256, 256, 0, stream>>>(hidden, Wn, bn, out_coords);

    // flash-style gravity attention -> bf16 O
    gravity_attn<<<dim3(S_LEN / 128, H_NUM, B_DIM), 256, 0, stream>>>(
        zbuf, zsqbuf, Vbf, gamma, Obf);

    // output projection: hidden' = O @ Wo^T + bo  (f32 out)
    gemm_bf16<false><<<dim3(32, 16), 256, 0, stream>>>(
        Obf, Wobf, bo, (void*)out_hidden, B_DIM * S_LEN, D_DIM, D_DIM);
}
